// GraniteMoeParallelExperts_49435073577021
// MI455X (gfx1250) — matrix operand order repaired and checked by reading, not compiled, not run on hardware
//
#include <hip/hip_runtime.h>
#include <hip/hip_bf16.h>

// ---------------------------------------------------------------------------
// Grouped GEMM (MoE parallel experts) for gfx1250 (MI455X).
// out[m, n] = sum_k x[m, k] * w[e, n, k]  for the expert e owning row m.
//
// bf16x3 split-precision WMMA (v_wmma_f32_16x16x32_bf16), f32 accumulation:
//   x = x_hi + x_lo,  w = w_hi + w_lo   (hi = truncated bf16, lo = truncated
//   bf16 of the residual);  x*w ~= x_hi*w_hi + x_hi*w_lo + x_lo*w_hi
//   -> total error ~2^-15 relative, vs ~2^-8 for single-pass bf16.
// Roofline: 103 GFLOP / 587 MB compulsory bytes -> ~25 us memory floor at
// 23.3 TB/s; bf16x3 compute ~62 us vs ~165 us for native f32 WMMA (16x16x4).
// Split cost: 1 v_and + 1/2 v_pk_add_f32 + 1 v_perm_b32 per element
// (~2.5 VALU/float) -- no bf16 scalar moves, fully hidden under the WMMAs.
// ---------------------------------------------------------------------------

typedef __bf16 bf16_t;
typedef __attribute__((ext_vector_type(16))) bf16_t v16bf;
typedef __attribute__((ext_vector_type(8)))  bf16_t v8bf;
typedef __attribute__((ext_vector_type(8)))  float  v8f;

union Frag16 {
    v16bf v;
    v8bf  h[2];
};

static constexpr int K_DIM   = 1536;
static constexpr int N_DIM   = 1024;
static constexpr int NUM_E   = 40;
static constexpr int BM      = 128;
static constexpr int BN      = 128;
static constexpr int BK      = 32;
static constexpr int THREADS = 256;
static constexpr int LDA     = 40;            // bf16 elems per LDS row (80B stride: conflict-free, 16B aligned)
static constexpr int KSTEPS  = K_DIM / BK;    // 48
static constexpr int MT      = 7;             // ceil(820/128) == ceil(819/128)
static constexpr int NT      = N_DIM / BN;    // 8

// Split float4 into bf16 hi (truncated) / lo (truncated residual); store 4 each.
// v_perm_b32 selector 0x07060302: result = { b.hi16, a.hi16 } (a=high dword).
__device__ __forceinline__ void split_store4(unsigned short* ph, unsigned short* pl,
                                             float4 v) {
    unsigned int u0 = __float_as_uint(v.x), u1 = __float_as_uint(v.y);
    unsigned int u2 = __float_as_uint(v.z), u3 = __float_as_uint(v.w);
    // hi value as f32: single AND per element
    float h0 = __uint_as_float(u0 & 0xFFFF0000u);
    float h1 = __uint_as_float(u1 & 0xFFFF0000u);
    float h2 = __uint_as_float(u2 & 0xFFFF0000u);
    float h3 = __uint_as_float(u3 & 0xFFFF0000u);
    // residuals (compiler fuses pairs into v_pk_add_f32 neg)
    unsigned int r0 = __float_as_uint(v.x - h0);
    unsigned int r1 = __float_as_uint(v.y - h1);
    unsigned int r2 = __float_as_uint(v.z - h2);
    unsigned int r3 = __float_as_uint(v.w - h3);
    uint2 hp, lp;
    hp.x = __builtin_amdgcn_perm(u1, u0, 0x07060302u);  // {u0.hi16, u1.hi16}
    hp.y = __builtin_amdgcn_perm(u3, u2, 0x07060302u);  // {u2.hi16, u3.hi16}
    lp.x = __builtin_amdgcn_perm(r1, r0, 0x07060302u);  // {r0.hi16, r1.hi16}
    lp.y = __builtin_amdgcn_perm(r3, r2, 0x07060302u);  // {r2.hi16, r3.hi16}
    *(uint2*)ph = hp;
    *(uint2*)pl = lp;
}

__global__ __launch_bounds__(THREADS)
void moe_grouped_gemm_bf16x3(const float* __restrict__ x,
                             const float* __restrict__ w,
                             const int*   __restrict__ expert_size,
                             float*       __restrict__ out) {
    // ---- block -> (expert, m-tile, n-tile); expert-major for L2 locality ----
    const int b  = blockIdx.x;
    const int e  = b / (MT * NT);
    const int r0 = b % (MT * NT);
    const int mt = r0 / NT;
    const int nt = r0 % NT;

    int off = 0;
    #pragma unroll 1
    for (int i = 0; i < e; ++i) off += expert_size[i];
    const int rows = expert_size[e];

    const int row0 = mt * BM;
    if (row0 >= rows) return;   // uniform per block: safe w.r.t. barriers

    const float* Ag = x + (size_t)off * K_DIM;                           // rows x K
    const float* Bg = w + ((size_t)e * N_DIM + (size_t)nt * BN) * K_DIM; // BN x K

    // ---- LDS staging: hi/lo bf16 tiles for A (128xBK) and B (128xBK) ----
    __shared__ unsigned short sAh[BM * LDA];
    __shared__ unsigned short sAl[BM * LDA];
    __shared__ unsigned short sBh[BN * LDA];
    __shared__ unsigned short sBl[BN * LDA];

    const int tid = threadIdx.x;
    const int lr  = tid >> 3;          // 0..31 (staging row group)
    const int lc  = (tid & 7) * 4;     // 0..28 (staging k column, float4)

    const int lane  = tid & 31;
    const int wv    = tid >> 5;        // 0..7
    const int wm    = wv & 3;          // 4 waves along M (32 rows each)
    const int wn    = wv >> 2;         // 2 waves along N (64 cols each)
    const int l16   = lane & 15;
    const int lhalf = lane >> 4;

    // ---- accumulators: 2 (M frags) x 4 (N frags) of 16x16 f32 ----
    v8f acc[2][4];
    #pragma unroll
    for (int i = 0; i < 2; ++i)
        #pragma unroll
        for (int j = 0; j < 4; ++j)
            acc[i][j] = (v8f){0.f, 0.f, 0.f, 0.f, 0.f, 0.f, 0.f, 0.f};

    // ---- ping-pong register prefetch buffers (no end-of-iter copies) ----
    float4 aReg[2][4], bReg[2][4];

    #pragma unroll
    for (int t = 0; t < 4; ++t) {
        const int r = lr + t * 32;
        aReg[0][t] = (row0 + r < rows)
                         ? *(const float4*)(Ag + (size_t)(row0 + r) * K_DIM + lc)
                         : make_float4(0.f, 0.f, 0.f, 0.f);
        bReg[0][t] = *(const float4*)(Bg + (size_t)r * K_DIM + lc);
    }

    #pragma unroll 2
    for (int kk = 0; kk < KSTEPS; ++kk) {
        const int cur = kk & 1;
        const int nxt = cur ^ 1;

        __syncthreads();   // previous compute finished reading LDS

        // stage current tile (f32 -> bf16 hi/lo) into LDS
        #pragma unroll
        for (int t = 0; t < 4; ++t) {
            const int r = lr + t * 32;
            split_store4(&sAh[r * LDA + lc], &sAl[r * LDA + lc], aReg[cur][t]);
            split_store4(&sBh[r * LDA + lc], &sBl[r * LDA + lc], bReg[cur][t]);
        }

        __syncthreads();   // LDS tile visible to all waves

        // issue next k-step's global loads early: VMEM overlaps the WMMAs
        if (kk + 1 < KSTEPS) {
            const int k0 = (kk + 1) * BK;
            #pragma unroll
            for (int t = 0; t < 4; ++t) {
                const int r = lr + t * 32;
                aReg[nxt][t] = (row0 + r < rows)
                                   ? *(const float4*)(Ag + (size_t)(row0 + r) * K_DIM + k0 + lc)
                                   : make_float4(0.f, 0.f, 0.f, 0.f);
                bReg[nxt][t] = *(const float4*)(Bg + (size_t)r * K_DIM + k0 + lc);
            }
        }

        // ---- fragments from LDS (ISA §7.12.2 layouts) ----
        // A (16x32): lane holds M=l16, K in [c,c+8) u [c+16,c+24), c=lhalf*8
        Frag16 ah[2], al[2];
        #pragma unroll
        for (int i = 0; i < 2; ++i) {
            const int rowA = wm * 32 + i * 16 + l16;
            const unsigned short* ph = &sAh[rowA * LDA + lhalf * 8];
            const unsigned short* pl = &sAl[rowA * LDA + lhalf * 8];
            ah[i].h[0] = *(const v8bf*)(ph);
            ah[i].h[1] = *(const v8bf*)(ph + 16);
            al[i].h[0] = *(const v8bf*)(pl);
            al[i].h[1] = *(const v8bf*)(pl + 16);
        }
        // B (32x16): lane holds N=l16, K in [c,c+8) u [c+16,c+24), c=lhalf*8
        Frag16 bh[4], bl[4];
        #pragma unroll
        for (int j = 0; j < 4; ++j) {
            const int rowB = wn * 64 + j * 16 + l16;
            const unsigned short* ph = &sBh[rowB * LDA + lhalf * 8];
            const unsigned short* pl = &sBl[rowB * LDA + lhalf * 8];
            bh[j].h[0] = *(const v8bf*)(ph);
            bh[j].h[1] = *(const v8bf*)(ph + 16);
            bl[j].h[0] = *(const v8bf*)(pl);
            bl[j].h[1] = *(const v8bf*)(pl + 16);
        }

        // ---- 2x4 tiles x 3 split products: 24 v_wmma_f32_16x16x32_bf16 ----
        #pragma unroll
        for (int i = 0; i < 2; ++i) {
            #pragma unroll
            for (int j = 0; j < 4; ++j) {
                acc[i][j] = __builtin_amdgcn_wmma_f32_16x16x32_bf16(
                    false, ah[i].v, false, bh[j].v, (short)0, acc[i][j], false, false);
                acc[i][j] = __builtin_amdgcn_wmma_f32_16x16x32_bf16(
                    false, ah[i].v, false, bl[j].v, (short)0, acc[i][j], false, false);
                acc[i][j] = __builtin_amdgcn_wmma_f32_16x16x32_bf16(
                    false, al[i].v, false, bh[j].v, (short)0, acc[i][j], false, false);
            }
        }
    }

    // ---- store C/D: 16x16 f32 layout (VGPR g: M=g+8*lhalf, N=l16) ----
    float* Cg = out + (size_t)off * N_DIM;
    #pragma unroll
    for (int i = 0; i < 2; ++i) {
        const int mBase = row0 + wm * 32 + i * 16 + lhalf * 8;
        #pragma unroll
        for (int j = 0; j < 4; ++j) {
            const int n = nt * BN + wn * 64 + j * 16 + l16;
            #pragma unroll
            for (int g = 0; g < 8; ++g) {
                const int m = mBase + g;
                if (m < rows) Cg[(size_t)m * N_DIM + n] = acc[i][j][g];
            }
        }
    }
}

extern "C" void kernel_launch(void* const* d_in, const int* in_sizes, int n_in,
                              void* d_out, int out_size, void* d_ws, size_t ws_size,
                              hipStream_t stream) {
    const float* x   = (const float*)d_in[0];  // [32768, 1536] f32
    const float* w   = (const float*)d_in[1];  // [40, 1024, 1536] f32
    const int*   es  = (const int*)d_in[2];    // [40] int32
    float*       out = (float*)d_out;          // [32768, 1024] f32

    const int grid = NUM_E * MT * NT;          // 40 * 7 * 8 = 2240 blocks
    moe_grouped_gemm_bf16x3<<<grid, THREADS, 0, stream>>>(x, w, es, out);
}
